// EncodeProcessDecode_1554778161263
// MI455X (gfx1250) — matrix operand order repaired and checked by reading, not compiled, not run on hardware
//
#include <hip/hip_runtime.h>
#include <hip/hip_bf16.h>

// ---------------------------------------------------------------------------
// EncodeProcessDecode GNN for gfx1250 (MI455X), f16 WMMA w/ f32 accumulation.
// Templated per-stage kernels: no runtime mode branches, fully unrolled K.
// ---------------------------------------------------------------------------

typedef __attribute__((ext_vector_type(16))) _Float16 v16h;
typedef __attribute__((ext_vector_type(8)))  _Float16 v8h;
typedef __attribute__((ext_vector_type(8)))  float    v8f;

#define HSTRIDE 136          // LDS row stride in halves (16B aligned, dodges bank conflicts)
#define WPB     4            // waves per block (wave32)
#define GNN_LN_EPS 1e-5f

__device__ __forceinline__ v16h combine16(v8h lo, v8h hi) {
  union { v16h v; v8h h[2]; } u;
  u.h[0] = lo; u.h[1] = hi;
  return u.v;
}

// A fragment (16x32 f16, row-major source): lane m=lane&15, half=lane>>4.
// elements 0..7  -> K = kOff + 8*half + [0..8)
// elements 8..15 -> K = kOff + 16 + 8*half + [0..8)
__device__ __forceinline__ v16h load_a_f16(const _Float16* rowptr, int kOff, int half) {
  const _Float16* p = rowptr + kOff + 8 * half;
  return combine16(*(const v8h*)p, *(const v8h*)(p + 16));
}

__device__ __forceinline__ v16h load_a_f32(const float* rowptr, int kOff, int half) {
  const float* p = rowptr + kOff + 8 * half;
  v8h lo, hi;
#pragma unroll
  for (int i = 0; i < 8; ++i) { lo[i] = (_Float16)p[i]; hi[i] = (_Float16)p[16 + i]; }
  return combine16(lo, hi);
}

__device__ __forceinline__ v8f wmma_f16(v16h a, v16h b, v8f c) {
  // v_wmma_f32_16x16x32_f16  (neg_a, A, neg_b, B, c_mod, C, reuse_a, reuse_b)
  return __builtin_amdgcn_wmma_f32_16x16x32_f16(false, a, false, b, (short)0, c, false, false);
}

// ---------------------------------------------------------------------------
// Fused 3-layer MLP (+ optional LayerNorm / residual / scatter-add / decode)
// MODE: 0=encoder (LN, plain write)  1=proc-edge (gather+LN+residual+scatter)
//       2=proc-node (agg input+LN+residual)  3=decoder (3-col f32 output)
// Each wave computes 16 rows x 128 cols; inter-layer transpose via LDS.
// ---------------------------------------------------------------------------
template <int MODE, int K0T>
__global__ __launch_bounds__(WPB * 32)
void gnn_mlp(int nrows,
             const _Float16* __restrict__ in0, int in0_stride,
             const _Float16* __restrict__ el16,
             const float* __restrict__ aggIn,
             const int* __restrict__ senders, const int* __restrict__ receivers,
             const _Float16* __restrict__ W0t, const _Float16* __restrict__ W1t,
             const _Float16* __restrict__ W2t,
             const float* __restrict__ b0, const float* __restrict__ b1,
             const float* __restrict__ b2,
             const float* __restrict__ gmm, const float* __restrict__ bta,
             float* __restrict__ out32, _Float16* __restrict__ out16,
             float* __restrict__ aggOut, float* __restrict__ decOut) {
  __shared__ __align__(16) _Float16 lds[WPB][16 * HSTRIDE];

  const int wave = threadIdx.x >> 5;
  const int lane = threadIdx.x & 31;
  const int nb   = lane & 15;   // C-tile column / A row within block
  const int half = lane >> 4;
  int rowBase = (blockIdx.x * WPB + wave) << 4;
  const bool active = rowBase < nrows;     // nrows % 16 == 0 for this model
  if (!active) rowBase = 0;                // clamp for safe (discarded) loads
  const int arow = rowBase + nb;

  int srow = arow, rrow = arow;
  if (MODE == 1) { srow = senders[arow]; rrow = receivers[arow]; }

  _Float16* hbuf = &lds[wave][0];
  v8f acc[8];

  // ---------------- layer 1 ----------------
#pragma unroll
  for (int t = 0; t < 8; ++t)
#pragma unroll
    for (int i = 0; i < 8; ++i) acc[t][i] = 0.f;

#pragma unroll
  for (int kt = 0; kt < K0T; ++kt) {
    v16h a;
    if (MODE == 1) {                       // concat [xl[s] | xl[r] | el]
      if (kt < 4)      a = load_a_f16(in0  + (size_t)srow * 128, kt * 32, half);
      else if (kt < 8) a = load_a_f16(in0  + (size_t)rrow * 128, (kt - 4) * 32, half);
      else             a = load_a_f16(el16 + (size_t)arow * 128, (kt - 8) * 32, half);
    } else if (MODE == 2) {                // concat [xl | agg(f32)]
      if (kt < 4)      a = load_a_f16(in0   + (size_t)arow * 128, kt * 32, half);
      else             a = load_a_f32(aggIn + (size_t)arow * 128, (kt - 4) * 32, half);
    } else {
      a = load_a_f16(in0 + (size_t)arow * in0_stride, kt * 32, half);
    }
    const _Float16* wk = W0t + (size_t)kt * (8 * 512);
    if (kt + 1 < K0T)
      __builtin_prefetch(W0t + (size_t)(kt + 1) * (8 * 512) + lane * 128, 0, 0);
#pragma unroll
    for (int nt = 0; nt < 8; ++nt) {
      v16h b = *(const v16h*)(wk + nt * 512 + lane * 16);
      acc[nt] = wmma_f16(a, b, acc[nt]);
    }
  }
#pragma unroll
  for (int nt = 0; nt < 8; ++nt) {
    float bv = b0[nt * 16 + nb];
#pragma unroll
    for (int i = 0; i < 8; ++i) {
      float v = acc[nt][i] + bv;
      hbuf[(i + 8 * half) * HSTRIDE + nt * 16 + nb] = (_Float16)(v > 0.f ? v : 0.f);
    }
  }
  __syncthreads();

  // ---------------- layer 2 ----------------
#pragma unroll
  for (int t = 0; t < 8; ++t)
#pragma unroll
    for (int i = 0; i < 8; ++i) acc[t][i] = 0.f;
#pragma unroll
  for (int kt = 0; kt < 4; ++kt) {
    v16h a = load_a_f16(hbuf + nb * HSTRIDE, kt * 32, half);
#pragma unroll
    for (int nt = 0; nt < 8; ++nt) {
      v16h b = *(const v16h*)(W1t + (size_t)(kt * 8 + nt) * 512 + lane * 16);
      acc[nt] = wmma_f16(a, b, acc[nt]);
    }
  }
  __syncthreads();
#pragma unroll
  for (int nt = 0; nt < 8; ++nt) {
    float bv = b1[nt * 16 + nb];
#pragma unroll
    for (int i = 0; i < 8; ++i) {
      float v = acc[nt][i] + bv;
      hbuf[(i + 8 * half) * HSTRIDE + nt * 16 + nb] = (_Float16)(v > 0.f ? v : 0.f);
    }
  }
  __syncthreads();

  // ---------------- layer 3 ----------------
  constexpr int NT3 = (MODE == 3) ? 1 : 8;   // decoder: 128->3 (padded to 16)
#pragma unroll
  for (int t = 0; t < 8; ++t)
#pragma unroll
    for (int i = 0; i < 8; ++i) acc[t][i] = 0.f;
#pragma unroll
  for (int kt = 0; kt < 4; ++kt) {
    v16h a = load_a_f16(hbuf + nb * HSTRIDE, kt * 32, half);
#pragma unroll
    for (int nt = 0; nt < NT3; ++nt) {
      v16h b = *(const v16h*)(W2t + (size_t)(kt * NT3 + nt) * 512 + lane * 16);
      acc[nt] = wmma_f16(a, b, acc[nt]);
    }
  }

  if (MODE == 3) {                        // decoder output: [N,3] f32
    float bv = (nb < 3) ? b2[nb] : 0.f;
    if (active && nb < 3) {
#pragma unroll
      for (int i = 0; i < 8; ++i)
        decOut[(size_t)(rowBase + i + 8 * half) * 3 + nb] = acc[0][i] + bv;
    }
    return;
  }

#pragma unroll
  for (int nt = 0; nt < 8; ++nt) {
    float bv = b2[nt * 16 + nb];
#pragma unroll
    for (int i = 0; i < 8; ++i) acc[nt][i] += bv;
  }

  // ------- LayerNorm: per-row (i,half) reduction across the 16-lane half ----
  float mu[8], rs[8];
#pragma unroll
  for (int i = 0; i < 8; ++i) {
    float su = 0.f, sq = 0.f;
#pragma unroll
    for (int nt = 0; nt < 8; ++nt) { float v = acc[nt][i]; su += v; sq += v * v; }
#pragma unroll
    for (int mk = 1; mk < 16; mk <<= 1) {
      su += __shfl_xor(su, mk, 32);
      sq += __shfl_xor(sq, mk, 32);
    }
    float m_ = su * (1.f / 128.f);
    mu[i] = m_;
    rs[i] = rsqrtf(sq * (1.f / 128.f) - m_ * m_ + GNN_LN_EPS);
  }

  int ridx[8];
#pragma unroll
  for (int i = 0; i < 8; ++i) ridx[i] = 0;
  if (MODE == 1 && active) {
#pragma unroll
    for (int i = 0; i < 8; ++i) ridx[i] = receivers[rowBase + i + 8 * half];
  }

  // ------- epilogue: LN * gamma + beta, residual, scatter-add ---------------
#pragma unroll
  for (int nt = 0; nt < 8; ++nt) {
    int col = nt * 16 + nb;
    float g = gmm[col], bb = bta[col];
#pragma unroll
    for (int i = 0; i < 8; ++i) {
      float y = (acc[nt][i] - mu[i]) * rs[i] * g + bb;
      if (active) {
        size_t r = (size_t)(rowBase + i + 8 * half);
        float o = (MODE == 0) ? y : (out32[r * 128 + col] + y);
        out32[r * 128 + col] = o;
        out16[r * 128 + col] = (_Float16)o;
        if (MODE == 1) atomicAdd(&aggOut[(size_t)ridx[i] * 128 + col], y);
      }
    }
  }
}

// ---------------------------------------------------------------------------
// Prep kernels
// ---------------------------------------------------------------------------
// Pack f32 weight [K][N] into zero-padded, WMMA-B-swizzled f16 tiles:
// tile (kt,nt) = 512 halves; lane = (n&15) + 16*((k>>3)&1); elem = (k&7) + 8*((k&31)>>4).
__global__ void pack_w_kernel(const float* __restrict__ W, _Float16* __restrict__ dst,
                              int K, int N, int Kpad, int Npad) {
  int idx = blockIdx.x * blockDim.x + threadIdx.x;
  int tot = Kpad * Npad;
  if (idx >= tot) return;
  int k = idx / Npad, n = idx % Npad;
  float v = (k < K && n < N) ? W[(size_t)k * N + n] : 0.f;
  int kt = k >> 5, nt = n >> 4, kl = k & 31;
  int hf = (kl >> 3) & 1, e = (kl & 7) + ((kl >> 4) << 3);
  int ln = (n & 15) + (hf << 4);
  int NT = Npad >> 4;
  dst[((size_t)(kt * NT + nt) * 32 + ln) * 16 + e] = (_Float16)v;
}

__global__ void pad_rows_kernel(const float* __restrict__ src, _Float16* __restrict__ dst,
                                int rows, int K, int Kpad) {
  int idx = blockIdx.x * blockDim.x + threadIdx.x;
  if (idx >= rows * Kpad) return;
  int r = idx / Kpad, k = idx % Kpad;
  dst[idx] = (k < K) ? (_Float16)src[(size_t)r * K + k] : (_Float16)0.f;
}

__global__ void zero_kernel(float* __restrict__ p, size_t n) {
  size_t i = (size_t)blockIdx.x * blockDim.x + threadIdx.x;
  size_t stride = (size_t)gridDim.x * blockDim.x;
  for (; i < n; i += stride) p[i] = 0.f;
}

// ---------------------------------------------------------------------------
// Host launcher
// ---------------------------------------------------------------------------
extern "C" void kernel_launch(void* const* d_in, const int* in_sizes, int n_in,
                              void* d_out, int out_size, void* d_ws, size_t ws_size,
                              hipStream_t stream) {
  (void)out_size; (void)ws_size;
  const int NODE_IN = 30, EDGE_IN = 4;
  const float* X  = (const float*)d_in[0];
  const float* EA = (const float*)d_in[1];
  const int N = in_sizes[0] / NODE_IN;
  const int E = in_sizes[1] / EDGE_IN;

  // --- resolve the 102 parameter leaves (insertion-order pytree flatten) ---
  const float* leaf[102];
  const int* SND; const int* RCV;
  if (n_in >= 106) {                 // one d_in entry per leaf
    for (int j = 0; j < 102; ++j) leaf[j] = (const float*)d_in[2 + j];
    SND = (const int*)d_in[104]; RCV = (const int*)d_in[105];
  } else {                           // params as a single flat f32 blob
    const float* base = (const float*)d_in[2];
    int sizes[102]; int j = 0;
    auto mlpln = [&](int k0) {
      sizes[j++] = k0 * 128; sizes[j++] = 128;
      sizes[j++] = 128 * 128; sizes[j++] = 128;
      sizes[j++] = 128 * 128; sizes[j++] = 128;
      sizes[j++] = 128; sizes[j++] = 128;
    };
    mlpln(30); mlpln(4);
    for (int s = 0; s < 5; ++s) { mlpln(384); mlpln(256); }
    sizes[j++] = 128 * 128; sizes[j++] = 128;
    sizes[j++] = 128 * 128; sizes[j++] = 128;
    sizes[j++] = 128 * 3;   sizes[j++] = 3;
    size_t off = 0;
    for (int t = 0; t < 102; ++t) { leaf[t] = base + off; off += sizes[t]; }
    SND = (const int*)d_in[3]; RCV = (const int*)d_in[4];
  }

  // --- carve workspace ---
  char* wp = (char*)d_ws;
  auto carve = [&](size_t bytes) -> void* {
    void* r = (void*)wp;
    wp += (bytes + 255) & ~(size_t)255;
    return r;
  };
  float*    xl32 = (float*)carve((size_t)N * 128 * 4);
  _Float16* xl16 = (_Float16*)carve((size_t)N * 128 * 2);
  float*    el32 = (float*)carve((size_t)E * 128 * 4);
  _Float16* el16 = (_Float16*)carve((size_t)E * 128 * 2);
  float*    agg  = (float*)carve((size_t)N * 128 * 4);
  _Float16* x16  = (_Float16*)carve((size_t)N * 32 * 2);
  _Float16* ea16 = (_Float16*)carve((size_t)E * 32 * 2);

  auto packW = [&](const float* src, int K, int Nn) -> _Float16* {
    int Kp = (K + 31) & ~31, Np = (Nn + 15) & ~15;
    _Float16* dst = (_Float16*)carve((size_t)Kp * Np * 2);
    int tot = Kp * Np;
    pack_w_kernel<<<(tot + 255) / 256, 256, 0, stream>>>(src, dst, K, Nn, Kp, Np);
    return dst;
  };

  struct MlpW { _Float16 *W0, *W1, *W2; const float *b0, *b1, *b2, *g, *bt; };
  auto getMlp = [&](int base, int K0) -> MlpW {
    MlpW m;
    m.W0 = packW(leaf[base + 0], K0, 128);  m.b0 = leaf[base + 1];
    m.W1 = packW(leaf[base + 2], 128, 128); m.b1 = leaf[base + 3];
    m.W2 = packW(leaf[base + 4], 128, 128); m.b2 = leaf[base + 5];
    m.g  = leaf[base + 6];                  m.bt = leaf[base + 7];
    return m;
  };
  MlpW encN = getMlp(0, 30);
  MlpW encE = getMlp(8, 4);
  MlpW procE[5], procN[5];
  for (int s = 0; s < 5; ++s) {
    procE[s] = getMlp(16 + 16 * s, 384);
    procN[s] = getMlp(24 + 16 * s, 256);
  }
  MlpW dec;
  dec.W0 = packW(leaf[96], 128, 128); dec.b0 = leaf[97];
  dec.W1 = packW(leaf[98], 128, 128); dec.b1 = leaf[99];
  dec.W2 = packW(leaf[100], 128, 3);  dec.b2 = leaf[101];
  dec.g = nullptr; dec.bt = nullptr;

  pad_rows_kernel<<<(N * 32 + 255) / 256, 256, 0, stream>>>(X, x16, N, 30, 32);
  pad_rows_kernel<<<(E * 32 + 255) / 256, 256, 0, stream>>>(EA, ea16, E, 4, 32);

  auto nblocks = [](int rows) { return (rows + 16 * WPB - 1) / (16 * WPB); };

  // encode
  gnn_mlp<0, 1><<<nblocks(N), WPB * 32, 0, stream>>>(
      N, x16, 32, nullptr, nullptr, SND, RCV,
      encN.W0, encN.W1, encN.W2, encN.b0, encN.b1, encN.b2, encN.g, encN.bt,
      xl32, xl16, nullptr, nullptr);
  gnn_mlp<0, 1><<<nblocks(E), WPB * 32, 0, stream>>>(
      E, ea16, 32, nullptr, nullptr, SND, RCV,
      encE.W0, encE.W1, encE.W2, encE.b0, encE.b1, encE.b2, encE.g, encE.bt,
      el32, el16, nullptr, nullptr);

  // process: 5 InteractionNetwork steps
  for (int s = 0; s < 5; ++s) {
    zero_kernel<<<512, 256, 0, stream>>>(agg, (size_t)N * 128);
    gnn_mlp<1, 12><<<nblocks(E), WPB * 32, 0, stream>>>(
        E, xl16, 128, el16, nullptr, SND, RCV,
        procE[s].W0, procE[s].W1, procE[s].W2,
        procE[s].b0, procE[s].b1, procE[s].b2, procE[s].g, procE[s].bt,
        el32, el16, agg, nullptr);
    gnn_mlp<2, 8><<<nblocks(N), WPB * 32, 0, stream>>>(
        N, xl16, 128, nullptr, agg, SND, RCV,
        procN[s].W0, procN[s].W1, procN[s].W2,
        procN[s].b0, procN[s].b1, procN[s].b2, procN[s].g, procN[s].bt,
        xl32, xl16, nullptr, nullptr);
  }

  // decode
  gnn_mlp<3, 4><<<nblocks(N), WPB * 32, 0, stream>>>(
      N, xl16, 128, nullptr, nullptr, SND, RCV,
      dec.W0, dec.W1, dec.W2, dec.b0, dec.b1, dec.b2, nullptr, nullptr,
      nullptr, nullptr, nullptr, (float*)d_out);
}
